// CausalMultiHeadSelfAttention_59433757442761
// MI455X (gfx1250) — matrix operand order repaired and checked by reading, not compiled, not run on hardware
//
#include <hip/hip_runtime.h>
#include <hip/hip_bf16.h>

#define B_  2
#define S_  2048
#define D_  1024
#define H_  16
#define DK_ 64
#define THETA_ 10000.0f

typedef __bf16 bf16_t;
typedef bf16_t v8bf  __attribute__((ext_vector_type(8)));
typedef bf16_t v16bf __attribute__((ext_vector_type(16)));
typedef float  v8f   __attribute__((ext_vector_type(8)));

// ---------- fragment loaders (wave32, per CDNA5 ISA 7.12.2 layouts) ----------

// A fragment: 16x32 (MxK) bf16, row-major source with row stride ld (elements).
// Lane L: m = L&15, base_k = (L>>4)*8; halves 0..7 -> K=base_k..base_k+7,
// halves 8..15 -> K=base_k+16..base_k+23.  Two contiguous 16B loads.
__device__ __forceinline__ v16bf load_a_frag(const bf16_t* base, int ld) {
    int lane = threadIdx.x & 31;
    int m  = lane & 15;
    int kb = (lane >> 4) * 8;
    const bf16_t* row = base + (size_t)m * ld;
    v8bf lo = *(const v8bf*)(row + kb);
    v8bf hi = *(const v8bf*)(row + kb + 16);
    return __builtin_shufflevector(lo, hi, 0,1,2,3,4,5,6,7,8,9,10,11,12,13,14,15);
}

// B fragment: 32x16 (KxN) bf16 where we hold B^T row-major (row n has the K run),
// row stride ld. Lane L: n = L&15, base_k = (L>>4)*8; halves 0..7 -> K=base_k..base_k+7,
// halves 8..15 -> K=base_k+16..base_k+23.
__device__ __forceinline__ v16bf load_bt_frag(const bf16_t* bt, int ld) {
    int lane = threadIdx.x & 31;
    int n  = lane & 15;
    int kb = (lane >> 4) * 8;
    const bf16_t* p = bt + (size_t)n * ld + kb;
    v8bf lo = *(const v8bf*)(p);
    v8bf hi = *(const v8bf*)(p + 16);
    return __builtin_shufflevector(lo, hi, 0,1,2,3,4,5,6,7,8,9,10,11,12,13,14,15);
}

__device__ __forceinline__ v8f wmma_bf16(v16bf a, v16bf b, v8f c) {
    return __builtin_amdgcn_wmma_f32_16x16x32_bf16(false, a, false, b, (short)0, c, false, false);
}

// Pipelined 32(M) x 64(N) x K tile accumulation with in-place ping-pong
// buffers (no cross-buffer register copies).  K must be a multiple of 64.
__device__ __forceinline__ void gemm_tile_acc(const bf16_t* __restrict__ X,
                                              const bf16_t* __restrict__ W,
                                              int m0, int n0, int K,
                                              v8f acc[2][4]) {
    const bf16_t* Xr0 = X + (size_t)m0 * K;
    const bf16_t* Xr1 = X + (size_t)(m0 + 16) * K;
    const bf16_t* Wr[4];
    #pragma unroll
    for (int t = 0; t < 4; ++t) Wr[t] = W + (size_t)(n0 + 16 * t) * K;

    v16bf a0[2], a1[2], b0[4], b1[4];
    a0[0] = load_a_frag(Xr0, K);
    a0[1] = load_a_frag(Xr1, K);
    #pragma unroll
    for (int t = 0; t < 4; ++t) b0[t] = load_bt_frag(Wr[t], K);
    a1[0] = load_a_frag(Xr0 + 32, K);
    a1[1] = load_a_frag(Xr1 + 32, K);
    #pragma unroll
    for (int t = 0; t < 4; ++t) b1[t] = load_bt_frag(Wr[t] + 32, K);

    for (int k = 0; k < K - 64; k += 64) {
        #pragma unroll
        for (int t = 0; t < 4; ++t) {
            acc[0][t] = wmma_bf16(a0[0], b0[t], acc[0][t]);
            acc[1][t] = wmma_bf16(a0[1], b0[t], acc[1][t]);
        }
        a0[0] = load_a_frag(Xr0 + k + 64, K);
        a0[1] = load_a_frag(Xr1 + k + 64, K);
        #pragma unroll
        for (int t = 0; t < 4; ++t) b0[t] = load_bt_frag(Wr[t] + k + 64, K);
        #pragma unroll
        for (int t = 0; t < 4; ++t) {
            acc[0][t] = wmma_bf16(a1[0], b1[t], acc[0][t]);
            acc[1][t] = wmma_bf16(a1[1], b1[t], acc[1][t]);
        }
        a1[0] = load_a_frag(Xr0 + k + 96, K);
        a1[1] = load_a_frag(Xr1 + k + 96, K);
        #pragma unroll
        for (int t = 0; t < 4; ++t) b1[t] = load_bt_frag(Wr[t] + k + 96, K);
    }
    #pragma unroll
    for (int t = 0; t < 4; ++t) {
        acc[0][t] = wmma_bf16(a0[0], b0[t], acc[0][t]);
        acc[1][t] = wmma_bf16(a0[1], b0[t], acc[1][t]);
    }
    #pragma unroll
    for (int t = 0; t < 4; ++t) {
        acc[0][t] = wmma_bf16(a1[0], b1[t], acc[0][t]);
        acc[1][t] = wmma_bf16(a1[1], b1[t], acc[1][t]);
    }
}

// ---------- kernels ----------

__global__ void cvt_f32_to_bf16(const float* __restrict__ src,
                                bf16_t* __restrict__ dst, int n) {
    int i = blockIdx.x * blockDim.x + threadIdx.x;
    if (i < n) dst[i] = (bf16_t)src[i];
}

// C[m, n] = sum_k X[m,k] * W[n,k]; bf16 output.  grid.z selects {Wq,Wk,Wv}.
__global__ __launch_bounds__(32)
void gemm_proj(const bf16_t* __restrict__ X, const bf16_t* __restrict__ Wbase,
               bf16_t* __restrict__ Ybase) {
    const int M = B_ * S_, K = D_, N = D_;
    const bf16_t* W = Wbase + (size_t)blockIdx.z * N * K;
    bf16_t* Y       = Ybase + (size_t)blockIdx.z * M * N;
    int m0 = blockIdx.x * 32;
    int n0 = blockIdx.y * 64;
    v8f acc[2][4] = {};
    gemm_tile_acc(X, W, m0, n0, K, acc);
    int lane = threadIdx.x & 31;
    int n = lane & 15, hi = lane >> 4;
    #pragma unroll
    for (int r = 0; r < 2; ++r)
        #pragma unroll
        for (int t = 0; t < 4; ++t)
            #pragma unroll
            for (int v = 0; v < 8; ++v)
                Y[(size_t)(m0 + 16 * r + v + hi * 8) * N + n0 + 16 * t + n] =
                    (bf16_t)acc[r][t][v];
}

// Same GEMM but fp32 output (final projection).
__global__ __launch_bounds__(32)
void gemm_out(const bf16_t* __restrict__ X, const bf16_t* __restrict__ W,
              float* __restrict__ Y) {
    const int M = B_ * S_, K = D_, N = D_;
    int m0 = blockIdx.x * 32;
    int n0 = blockIdx.y * 64;
    v8f acc[2][4] = {};
    gemm_tile_acc(X, W, m0, n0, K, acc);
    int lane = threadIdx.x & 31;
    int n = lane & 15, hi = lane >> 4;
    #pragma unroll
    for (int r = 0; r < 2; ++r)
        #pragma unroll
        for (int t = 0; t < 4; ++t)
            #pragma unroll
            for (int v = 0; v < 8; ++v)
                Y[(size_t)(m0 + 16 * r + v + hi * 8) * N + n0 + 16 * t + n] =
                    acc[r][t][v];
}

// RoPE + repack: Yq/Yk/Yv are bf16 [B,S,D] slabs (contiguous).
// Qb,Kb -> [B,H,S,DK] bf16 (rope applied, Qb scaled by 1/sqrt(DK));
// Vt    -> [B,H,DK,S] bf16 (transposed for contiguous B-fragments).
__global__ void rope_pack(const bf16_t* __restrict__ Y,
                          bf16_t* __restrict__ Qb, bf16_t* __restrict__ Kb,
                          bf16_t* __restrict__ Vt) {
    int idx = blockIdx.x * blockDim.x + threadIdx.x;
    const int total = B_ * S_ * D_;
    if (idx >= total) return;
    int d = idx % D_;
    int s = (idx / D_) % S_;
    int b = idx / (D_ * S_);
    int h  = d / DK_;
    int dk = d % DK_;
    const bf16_t* Yq = Y;
    const bf16_t* Yk = Y + (size_t)total;
    const bf16_t* Yv = Y + (size_t)2 * total;
    float q = (float)Yq[idx], k = (float)Yk[idx];
    int pair = dk >> 1;
    float inv  = __expf(-(float)(2 * pair) * (__logf(THETA_) / (float)DK_));
    float ang  = (float)s * inv;
    float c = __cosf(ang), sn = __sinf(ang);
    float qo, ko;
    if (dk & 1) {
        float q2 = (float)Yq[idx - 1], k2 = (float)Yk[idx - 1];
        qo = q * c + q2 * sn;  ko = k * c + k2 * sn;
    } else {
        float q2 = (float)Yq[idx + 1], k2 = (float)Yk[idx + 1];
        qo = q * c - q2 * sn;  ko = k * c - k2 * sn;
    }
    size_t o = ((size_t)(b * H_ + h) * S_ + s) * DK_ + dk;
    Qb[o] = (bf16_t)(qo * 0.125f);   // fold 1/sqrt(64)
    Kb[o] = (bf16_t)ko;
    Vt[((size_t)(b * H_ + h) * DK_ + dk) * S_ + s] = Yv[idx];
}

// Flash attention, one wave per (b,h, 16-query tile). Online softmax in the
// C-fragment layout; P goes through LDS to be re-read in the A layout.
// V fragments are fetched before the softmax VALU block so the loads
// overlap the exp/reduction math.
__global__ __launch_bounds__(32)
void flash_attn(const bf16_t* __restrict__ Qb, const bf16_t* __restrict__ Kb,
                const bf16_t* __restrict__ Vt, bf16_t* __restrict__ Ab) {
    __shared__ bf16_t pTile[16 * 32];
    int bh = blockIdx.y;             // b*H + h
    int q0 = blockIdx.x * 16;
    int lane = threadIdx.x & 31;
    int n = lane & 15, hi = lane >> 4;

    const bf16_t* Qh = Qb + (size_t)bh * S_ * DK_;
    const bf16_t* Kh = Kb + (size_t)bh * S_ * DK_;
    const bf16_t* Vh = Vt + (size_t)bh * DK_ * S_;

    v16bf aq0 = load_a_frag(Qh + (size_t)q0 * DK_, DK_);
    v16bf aq1 = load_a_frag(Qh + (size_t)q0 * DK_ + 32, DK_);

    v8f O[4] = {};
    float mrow[8], lrow[8];
    #pragma unroll
    for (int v = 0; v < 8; ++v) { mrow[v] = -1e30f; lrow[v] = 0.f; }

    const int kend = q0 + 16;        // causal: keys <= row < kend
    for (int j = 0; j < kend; j += 32) {
        v8f Sf[2] = {};
        #pragma unroll
        for (int t = 0; t < 2; ++t) {
            int kb = j + 16 * t;
            if (kb < kend) {
                v16bf b0 = load_bt_frag(Kh + (size_t)kb * DK_, DK_);
                v16bf b1 = load_bt_frag(Kh + (size_t)kb * DK_ + 32, DK_);
                Sf[t] = wmma_bf16(aq0, b0, Sf[t]);
                Sf[t] = wmma_bf16(aq1, b1, Sf[t]);
            }
        }
        // Hoist V fragment loads: independent of softmax, hides load latency.
        v16bf bv[4];
        #pragma unroll
        for (int t = 0; t < 4; ++t)
            bv[t] = load_bt_frag(Vh + (size_t)(16 * t) * S_ + j, S_);

        #pragma unroll
        for (int v = 0; v < 8; ++v) {
            int row = q0 + v + hi * 8;
            float e0 = Sf[0][v]; if (j + n      > row) e0 = -1e30f;
            float e1 = Sf[1][v]; if (j + 16 + n > row) e1 = -1e30f;
            float mx = fmaxf(e0, e1);
            #pragma unroll
            for (int sft = 1; sft < 16; sft <<= 1)
                mx = fmaxf(mx, __shfl_xor(mx, sft, 16));
            float mnew = fmaxf(mrow[v], mx);
            float scale = __expf(mrow[v] - mnew);
            float p0 = __expf(e0 - mnew);
            float p1 = __expf(e1 - mnew);
            float ps = p0 + p1;
            #pragma unroll
            for (int sft = 1; sft < 16; sft <<= 1)
                ps += __shfl_xor(ps, sft, 16);
            lrow[v] = lrow[v] * scale + ps;
            mrow[v] = mnew;
            #pragma unroll
            for (int t = 0; t < 4; ++t) O[t][v] *= scale;
            pTile[(v + hi * 8) * 32 + n]      = (bf16_t)p0;
            pTile[(v + hi * 8) * 32 + 16 + n] = (bf16_t)p1;
        }
        __syncthreads();
        // P (16x32) from LDS in the A layout: two contiguous ds_load_b128s.
        {
            int m  = lane & 15;
            int kb = (lane >> 4) * 8;
            v8bf lo = *(const v8bf*)(&pTile[m * 32 + kb]);
            v8bf hh = *(const v8bf*)(&pTile[m * 32 + kb + 16]);
            v16bf ap = __builtin_shufflevector(lo, hh,
                        0,1,2,3,4,5,6,7,8,9,10,11,12,13,14,15);
            #pragma unroll
            for (int t = 0; t < 4; ++t)
                O[t] = wmma_bf16(ap, bv[t], O[t]);
        }
        __syncthreads();
    }

    // epilogue: normalize and scatter to [B,S,D] with head interleave
    int b = bh / H_, h = bh % H_;
    #pragma unroll
    for (int v = 0; v < 8; ++v) {
        int row = q0 + v + hi * 8;
        float invl = 1.0f / lrow[v];
        #pragma unroll
        for (int t = 0; t < 4; ++t)
            Ab[((size_t)b * S_ + row) * D_ + h * DK_ + 16 * t + n] =
                (bf16_t)(O[t][v] * invl);
    }
}

// ---------- host ----------

extern "C" void kernel_launch(void* const* d_in, const int* in_sizes, int n_in,
                              void* d_out, int out_size, void* d_ws, size_t ws_size,
                              hipStream_t stream) {
    (void)in_sizes; (void)n_in; (void)out_size; (void)ws_size;
    const float* x  = (const float*)d_in[0];
    const float* Wq = (const float*)d_in[1];
    const float* Wk = (const float*)d_in[2];
    const float* Wv = (const float*)d_in[3];
    const float* Wo = (const float*)d_in[4];

    const size_t M   = (size_t)B_ * S_;      // 4096
    const size_t XE  = M * D_;               // 4M elements
    const size_t WE  = (size_t)D_ * D_;      // 1M elements

    bf16_t* xb = (bf16_t*)d_ws;                  // [M, D]
    bf16_t* wb = xb + XE;                        // Wq,Wk,Wv,Wo  (4 x [D,D])
    bf16_t* yb = wb + 4 * WE;                    // Q,K,V proj   (3 x [B,S,D])
    bf16_t* qb = yb + 3 * XE;                    // [B,H,S,DK]
    bf16_t* kb = qb + XE;                        // [B,H,S,DK]
    bf16_t* vt = kb + XE;                        // [B,H,DK,S]
    bf16_t* ab = vt + XE;                        // [B,S,D]

    const int T = 256;
    cvt_f32_to_bf16<<<(XE + T - 1) / T, T, 0, stream>>>(x, xb, (int)XE);
    cvt_f32_to_bf16<<<(WE + T - 1) / T, T, 0, stream>>>(Wq, wb + 0 * WE, (int)WE);
    cvt_f32_to_bf16<<<(WE + T - 1) / T, T, 0, stream>>>(Wk, wb + 1 * WE, (int)WE);
    cvt_f32_to_bf16<<<(WE + T - 1) / T, T, 0, stream>>>(Wv, wb + 2 * WE, (int)WE);
    cvt_f32_to_bf16<<<(WE + T - 1) / T, T, 0, stream>>>(Wo, wb + 3 * WE, (int)WE);

    // Q/K/V projections: grid (M/32, N/64, 3), one wave per block.
    gemm_proj<<<dim3(M / 32, D_ / 64, 3), 32, 0, stream>>>(xb, wb, yb);

    rope_pack<<<(XE + T - 1) / T, T, 0, stream>>>(yb, qb, kb, vt);

    // Attention: grid (S/16, B*H), one wave per block.
    flash_attn<<<dim3(S_ / 16, B_ * H_), 32, 0, stream>>>(qb, kb, vt, ab);

    // Output projection -> fp32 d_out.
    gemm_out<<<dim3(M / 32, D_ / 64), 32, 0, stream>>>(ab, wb + 3 * WE, (float*)d_out);
}
